// LocalState_3023656976979
// MI455X (gfx1250) — matrix operand order repaired and checked by reading, not compiled, not run on hardware
//
#include <hip/hip_runtime.h>
#include <hip/hip_bf16.h>
#include <math.h>

typedef __attribute__((ext_vector_type(16))) _Float16 v16h;
typedef __attribute__((ext_vector_type(8)))  float    v8f;

#define B_   4
#define C_   192
#define T_   2048
#define H_   4
#define HD_  48
#define HDP_ 64
#define JTOT_ 592   // 3*192 (Q,K,content) + 16 (decay)

#define KSTRIDE 72  // halves per K row in LDS (64 data + 8 pad -> bank spread)
#define VSTRIDE 40  // halves per V row in LDS (32 data + 8 pad)

// ---------------------------------------------------------------- helpers

__device__ __forceinline__ v16h frag_from(uint4 lo, uint4 hi) {
  union { uint4 u[2]; v16h v; } x;
  x.u[0] = lo; x.u[1] = hi;
  return x.v;
}

// A-matrix fragment (M=16 rows x K=32), 16-bit: lane holds 16 contiguous K:
//   K = kh*16 + [0..15]   (kh = lane>>4)
__device__ __forceinline__ v16h load_a32(const _Float16* row, int cbase, int kh) {
  const uint4 u0 = *(const uint4*)(row + cbase + kh * 16);
  const uint4 u1 = *(const uint4*)(row + cbase + kh * 16 + 8);
  return frag_from(u0, u1);
}

// B-matrix fragment (K=32 x N=16), 16-bit: lane n=L&15 holds 16 contiguous K:
//   K = kh*16 + [0..15]
__device__ __forceinline__ v16h load_b32(const _Float16* row, int cbase, int kh) {
  const uint4 u0 = *(const uint4*)(row + cbase + kh * 16);
  const uint4 u1 = *(const uint4*)(row + cbase + kh * 16 + 8);
  return frag_from(u0, u1);
}

__device__ __forceinline__ v8f wmma_f16(v16h a, v16h b, v8f c) {
  return __builtin_amdgcn_wmma_f32_16x16x32_f16(false, a, false, b, (short)0, c,
                                                false, false);
}

__device__ __forceinline__ uint4 shfl_xor16_u4(uint4 v) {
  uint4 r;
  r.x = (unsigned)__shfl_xor((int)v.x, 16, 32);
  r.y = (unsigned)__shfl_xor((int)v.y, 16, 32);
  r.z = (unsigned)__shfl_xor((int)v.z, 16, 32);
  r.w = (unsigned)__shfl_xor((int)v.w, 16, 32);
  return r;
}

// Async copy 16B: LDS[lds_off] = MEM[gaddr ... +15]; tracked with ASYNCcnt.
__device__ __forceinline__ void async_b128(uint32_t lds_off, const void* gaddr) {
  asm volatile("global_load_async_to_lds_b128 %0, %1, off"
               :: "v"(lds_off), "v"((uint64_t)(uintptr_t)gaddr)
               : "memory");
}

// ---------------------------------------------------------------- prep

__global__ void k_prep_weights(const float* __restrict__ Wq, const float* __restrict__ Wk,
                               const float* __restrict__ Wc, const float* __restrict__ Wd,
                               const float* __restrict__ Wp, const float* __restrict__ bq,
                               const float* __restrict__ bk, const float* __restrict__ bc,
                               const float* __restrict__ bd,
                               _Float16* __restrict__ Wt_all, _Float16* __restrict__ Wpt,
                               float* __restrict__ bcat) {
  int idx = blockIdx.x * 256 + threadIdx.x;
  const int NW1 = JTOT_ * C_;
  if (idx < NW1) {
    int j = idx / C_, c = idx % C_;
    float v;
    if (j < 192)      v = Wq[c * C_ + j];
    else if (j < 384) v = Wk[c * C_ + (j - 192)];
    else if (j < 576) v = Wc[c * C_ + (j - 384)];
    else              v = Wd[c * 16 + (j - 576)];
    Wt_all[j * C_ + c] = (_Float16)v;
  } else if (idx < NW1 + C_ * C_) {
    int r = idx - NW1;
    int j = r / C_, c = r % C_;
    Wpt[j * C_ + c] = (_Float16)Wp[c * C_ + j];
  }
  if (idx < JTOT_) {
    int j = idx;
    float bv;
    if (j < 192)      bv = bq[j];
    else if (j < 384) bv = bk[j - 192];
    else if (j < 576) bv = bc[j - 384];
    else              bv = bd[j - 576];
    bcat[j] = bv;
  }
}

// x (B,C,T) f32 -> xt (B,T,C) f16
__global__ void k_prep_x(const float* __restrict__ x, _Float16* __restrict__ xt) {
  int idx = blockIdx.x * 256 + threadIdx.x;
  if (idx >= B_ * C_ * T_) return;
  int b = idx / (C_ * T_);
  int rem = idx % (C_ * T_);
  int c = rem / T_, t = rem % T_;
  xt[((size_t)b * T_ + t) * C_ + c] = (_Float16)x[idx];
}

// zero the hd 48..63 pad columns of Qh and Kh
__global__ void k_zero_pad(_Float16* __restrict__ Qh, _Float16* __restrict__ Kh) {
  int idx = blockIdx.x * 256 + threadIdx.x;
  const int N = B_ * H_ * T_ * 16;
  if (idx >= 2 * N) return;
  int sel = idx / N, r = idx % N;
  int bht = r / 16, c = 48 + (r % 16);
  (sel ? Kh : Qh)[(size_t)bht * HDP_ + c] = (_Float16)0.f;
}

// ---------------------------------------------------------------- projections

// One wave computes a 16(t) x 16(j) tile of xt @ Wcat, K = 192 = 6 x 32.
__global__ __launch_bounds__(256) void k_proj(
    const _Float16* __restrict__ xt, const _Float16* __restrict__ Wt_all,
    const float* __restrict__ bcat, _Float16* __restrict__ Qh,
    _Float16* __restrict__ Kh, _Float16* __restrict__ Vt,
    float* __restrict__ dsum) {
  int lane = threadIdx.x & 31;
  int wid = blockIdx.x * 8 + (threadIdx.x >> 5);   // 0 .. B*128*37-1
  int jt = wid % 37;
  int tt = (wid / 37) & 127;
  int b  = wid / (37 * 128);
  int n = lane & 15, kh = lane >> 4;

  const _Float16* Arow = xt + ((size_t)b * T_ + tt * 16 + n) * C_;
  const _Float16* Brow = Wt_all + (size_t)(jt * 16 + n) * C_;
  v8f acc = {};
#pragma unroll
  for (int kk = 0; kk < 6; ++kk)
    acc = wmma_f16(load_a32(Arow, kk * 32, kh), load_b32(Brow, kk * 32, kh), acc);

  int j = jt * 16 + n;
  float bias = bcat[j];

  if (jt < 24) {  // Q (jt<12) or K
    _Float16* dst = (jt < 12) ? Qh : Kh;
    int jj = (jt < 12) ? j : j - 192;
    int h = jj / HD_, c = jj % HD_;
    size_t base = (size_t)(b * H_ + h) * T_ * HDP_ + c;
#pragma unroll
    for (int v = 0; v < 8; ++v) {
      int t = tt * 16 + 8 * kh + v;
      dst[base + (size_t)t * HDP_] = (_Float16)(acc[v] + bias);
    }
  } else if (jt < 36) {  // content -> Vt[b,h,c,T] (contiguous in t)
    int jj = j - 384;
    int h = jj / HD_, c = jj % HD_;
    union { _Float16 hh[8]; uint4 u; } pk;
#pragma unroll
    for (int v = 0; v < 8; ++v) pk.hh[v] = (_Float16)(acc[v] + bias);
    *(uint4*)(Vt + ((size_t)(b * H_ + h) * HD_ + c) * T_ + tt * 16 + 8 * kh) = pk.u;
  } else {  // decay: dsum[b,h,s] = sum_f (f+1)*sigmoid(y)/2
    int h = n >> 2;
#pragma unroll
    for (int v = 0; v < 8; ++v) {
      float y = acc[v] + bias;
      float f1 = (float)((n & 3) + 1);
      float e = f1 * 0.5f / (1.f + __expf(-y));
      e += __shfl_xor(e, 1, 32);
      e += __shfl_xor(e, 2, 32);
      if ((lane & 3) == 0)
        dsum[(size_t)(b * H_ + h) * T_ + tt * 16 + 8 * kh + v] = e;
    }
  }
}

// ---------------------------------------------------------------- flash attention

// Grid: (b,h) x 16 query blocks of 128; 8 waves/WG, one 16-query tile each.
// K/V tiles staged in LDS with async-to-LDS copies, double-buffered.
// Computes S^T tiles (lane = s, VGPR rows = t) so softmax-over-t is per-lane.
__global__ __launch_bounds__(256) void k_flash(
    const _Float16* __restrict__ Qh, const _Float16* __restrict__ Kh,
    const _Float16* __restrict__ Vt, const float* __restrict__ dsum,
    _Float16* __restrict__ resT) {
  __shared__ _Float16 sK[2][32 * KSTRIDE];
  __shared__ _Float16 sV[2][48 * VSTRIDE];

  int tid = threadIdx.x;
  int lane = tid & 31;
  int w = tid >> 5;
  int bh = blockIdx.x >> 4;
  int sb = blockIdx.x & 15;
  int b = bh >> 2, h = bh & 3;
  int n = lane & 15, kh = lane >> 4;
  int sL = sb * 128 + w * 16 + n;

  const _Float16* Qrow = Qh + ((size_t)bh * T_ + sL) * HDP_;
  v16h qb0 = load_b32(Qrow, 0, kh);
  v16h qb1 = load_b32(Qrow, 32, kh);
  float dsv = 0.5f * dsum[(size_t)bh * T_ + sL];  // bias = -|t-s| * dsv

  const _Float16* Kbase = Kh + (size_t)bh * T_ * HDP_;
  const _Float16* Vbase = Vt + (size_t)bh * HD_ * T_;

  // staging roles: K tile = 32 rows x 8 segs of 16B (all 256 threads),
  //                V tile = 48 rows x 4 segs of 16B (threads 0..191).
  int krow = tid >> 3, kseg = tid & 7;
  uint32_t kl0 = (uint32_t)(uintptr_t)&sK[0][krow * KSTRIDE + kseg * 8];
  uint32_t kl1 = (uint32_t)(uintptr_t)&sK[1][krow * KSTRIDE + kseg * 8];
  const _Float16* kg = Kbase + (size_t)krow * HDP_ + kseg * 8;
  int vrow = tid >> 2, vseg = tid & 3;
  uint32_t vl0 = (uint32_t)(uintptr_t)&sV[0][vrow * VSTRIDE + vseg * 8];
  uint32_t vl1 = (uint32_t)(uintptr_t)&sV[1][vrow * VSTRIDE + vseg * 8];
  const _Float16* vg = Vbase + (size_t)vrow * T_ + vseg * 8;

  const float scale = 0.14433756729740643f;  // 1/sqrt(48)
  float mrun = -1e30f, lrun = 0.f;
  v8f O0 = {}, O1 = {}, O2 = {};

  // prologue: stage chunk 0 into buffer 0
  async_b128(kl0, kg);
  if (tid < 192) async_b128(vl0, vg);

  for (int tc = 0; tc < T_ / 32; ++tc) {
    int buf = tc & 1;
    // prefetch next chunk into the other buffer (wraps on last iter; the
    // trailing barrier of tc-1 makes the overwrite safe).
    {
      int tn = (tc + 1) & 63;
      async_b128(buf ? kl0 : kl1, kg + (size_t)tn * 32 * HDP_);
      if (tid < 192) async_b128(buf ? vl0 : vl1, vg + tn * 32);
    }
    // wait for *this* chunk's async copies (in-order completion); waves 6,7
    // stage only K so their in-flight count differs (wave-uniform branch).
    if (w < 6) asm volatile("s_wait_asynccnt 2" ::: "memory");
    else       asm volatile("s_wait_asynccnt 1" ::: "memory");
    __syncthreads();

    int t0 = tc * 32;
    const _Float16* K0 = &sK[buf][(size_t)n * KSTRIDE];
    const _Float16* K1 = &sK[buf][(size_t)(16 + n) * KSTRIDE];
    v8f d0 = {}, d1 = {};
    d0 = wmma_f16(load_a32(K0, 0, kh), qb0, d0);
    d0 = wmma_f16(load_a32(K0, 32, kh), qb1, d0);
    d1 = wmma_f16(load_a32(K1, 0, kh), qb0, d1);
    d1 = wmma_f16(load_a32(K1, 32, kh), qb1, d1);

    float v0[8], v1[8];
    float cmax = -1e30f;
#pragma unroll
    for (int v = 0; v < 8; ++v) {
      int t = t0 + 8 * kh + v;
      float x0 = (t == sL) ? -100.f
                           : d0[v] * scale - fabsf((float)(t - sL)) * dsv;
      v0[v] = x0; cmax = fmaxf(cmax, x0);
      int t2 = t + 16;
      float x1 = (t2 == sL) ? -100.f
                            : d1[v] * scale - fabsf((float)(t2 - sL)) * dsv;
      v1[v] = x1; cmax = fmaxf(cmax, x1);
    }
    cmax = fmaxf(cmax, __shfl_xor(cmax, 16, 32));  // sync lane-pair stats
    float mnew = fmaxf(mrun, cmax);
    float corr = __expf(mrun - mnew);
    float lown = 0.f;
    union { _Float16 hh[8]; uint4 u; } p0, p1;
#pragma unroll
    for (int v = 0; v < 8; ++v) {
      float e0 = __expf(v0[v] - mnew);
      float e1 = __expf(v1[v] - mnew);
      lown += e0 + e1;
      p0.hh[v] = (_Float16)e0;
      p1.hh[v] = (_Float16)e1;
    }
    lrun = lrun * corr + lown;
    mrun = mnew;
#pragma unroll
    for (int v = 0; v < 8; ++v) { O0[v] *= corr; O1[v] *= corr; O2[v] *= corr; }

    // Build B-fragment of P' (K=t 32, N=s 16): exchange 8 exps across halves.
    uint4 send = kh ? p0.u : p1.u;
    uint4 recv = shfl_xor16_u4(send);
    v16h pb = kh ? frag_from(recv, p1.u) : frag_from(p0.u, recv);

    // O^T[c,s] += V^T x P' : A = sV rows (M=c), K = t-local chunk of 32.
    O0 = wmma_f16(load_a32(&sV[buf][(size_t)(0 + n) * VSTRIDE], 0, kh), pb, O0);
    O1 = wmma_f16(load_a32(&sV[buf][(size_t)(16 + n) * VSTRIDE], 0, kh), pb, O1);
    O2 = wmma_f16(load_a32(&sV[buf][(size_t)(32 + n) * VSTRIDE], 0, kh), pb, O2);

    __syncthreads();  // consumption done before this buffer is overwritten
  }

  float ltot = lrun + __shfl_xor(lrun, 16, 32);
  float inv = 1.f / ltot;
  size_t rbase = ((size_t)b * T_ + sL) * C_ + h * HD_;
  union { _Float16 hh[8]; uint4 u; } pk;
#pragma unroll
  for (int v = 0; v < 8; ++v) pk.hh[v] = (_Float16)(O0[v] * inv);
  *(uint4*)(resT + rbase + 0 + 8 * kh) = pk.u;
#pragma unroll
  for (int v = 0; v < 8; ++v) pk.hh[v] = (_Float16)(O1[v] * inv);
  *(uint4*)(resT + rbase + 16 + 8 * kh) = pk.u;
#pragma unroll
  for (int v = 0; v < 8; ++v) pk.hh[v] = (_Float16)(O2[v] * inv);
  *(uint4*)(resT + rbase + 32 + 8 * kh) = pk.u;
}

// ---------------------------------------------------------------- output proj

// out[b,c',t] = resT[b,t,:] @ Wp[:,c'] + bp[c'] + x[b,c',t]
__global__ __launch_bounds__(256) void k_final(
    const _Float16* __restrict__ resT, const _Float16* __restrict__ Wpt,
    const float* __restrict__ bp, const float* __restrict__ x,
    float* __restrict__ out) {
  int lane = threadIdx.x & 31;
  int wid = blockIdx.x * 8 + (threadIdx.x >> 5);  // 0 .. B*128*12-1
  int jt = wid % 12;
  int tt = (wid / 12) & 127;
  int b  = wid / (12 * 128);
  int n = lane & 15, kh = lane >> 4;

  const _Float16* Arow = resT + ((size_t)b * T_ + tt * 16 + n) * C_;
  const _Float16* Brow = Wpt + (size_t)(jt * 16 + n) * C_;
  v8f acc = {};
#pragma unroll
  for (int kk = 0; kk < 6; ++kk)
    acc = wmma_f16(load_a32(Arow, kk * 32, kh), load_b32(Brow, kk * 32, kh), acc);

  int j = jt * 16 + n;
  float bias = bp[j];
  size_t base = (size_t)(b * C_ + j) * T_ + tt * 16 + 8 * kh;
  const float4* xp = (const float4*)(x + base);
  float4* op = (float4*)(out + base);
  float4 xa = xp[0], xb = xp[1];
  float4 ra, rb;
  ra.x = acc[0] + bias + xa.x; ra.y = acc[1] + bias + xa.y;
  ra.z = acc[2] + bias + xa.z; ra.w = acc[3] + bias + xa.w;
  rb.x = acc[4] + bias + xb.x; rb.y = acc[5] + bias + xb.y;
  rb.z = acc[6] + bias + xb.z; rb.w = acc[7] + bias + xb.w;
  op[0] = ra; op[1] = rb;
}

// ---------------------------------------------------------------- launch

extern "C" void kernel_launch(void* const* d_in, const int* in_sizes, int n_in,
                              void* d_out, int out_size, void* d_ws, size_t ws_size,
                              hipStream_t stream) {
  (void)in_sizes; (void)n_in; (void)out_size; (void)ws_size;
  const float* x  = (const float*)d_in[0];
  const float* Wq = (const float*)d_in[1];
  const float* bq = (const float*)d_in[2];
  const float* Wk = (const float*)d_in[3];
  const float* bk = (const float*)d_in[4];
  const float* Wc = (const float*)d_in[5];
  const float* bc = (const float*)d_in[6];
  const float* Wd = (const float*)d_in[7];
  const float* bd = (const float*)d_in[8];
  const float* Wp = (const float*)d_in[9];
  const float* bp = (const float*)d_in[10];
  float* out = (float*)d_out;

  char* ws = (char*)d_ws;
  size_t cur = 0;
  auto take = [&](size_t bytes) -> void* {
    void* p = ws + cur;
    cur += (bytes + 255) & ~(size_t)255;
    return p;
  };
  _Float16* Wt_all = (_Float16*)take((size_t)JTOT_ * C_ * 2);
  _Float16* Wpt    = (_Float16*)take((size_t)C_ * C_ * 2);
  float*    bcat   = (float*)take((size_t)JTOT_ * 4);
  _Float16* xt     = (_Float16*)take((size_t)B_ * T_ * C_ * 2);
  _Float16* Qh     = (_Float16*)take((size_t)B_ * H_ * T_ * HDP_ * 2);
  _Float16* Kh     = (_Float16*)take((size_t)B_ * H_ * T_ * HDP_ * 2);
  _Float16* Vt     = (_Float16*)take((size_t)B_ * H_ * HD_ * T_ * 2);
  float*    dsum   = (float*)take((size_t)B_ * H_ * T_ * 4);
  _Float16* resT   = (_Float16*)take((size_t)B_ * T_ * C_ * 2);

  k_prep_weights<<<(JTOT_ * C_ + C_ * C_ + 255) / 256, 256, 0, stream>>>(
      Wq, Wk, Wc, Wd, Wp, bq, bk, bc, bd, Wt_all, Wpt, bcat);
  k_prep_x<<<(B_ * C_ * T_ + 255) / 256, 256, 0, stream>>>(x, xt);
  k_zero_pad<<<(2 * B_ * H_ * T_ * 16 + 255) / 256, 256, 0, stream>>>(Qh, Kh);
  k_proj<<<B_ * 128 * 37 / 8, 256, 0, stream>>>(xt, Wt_all, bcat, Qh, Kh, Vt, dsum);
  k_flash<<<B_ * H_ * 16, 256, 0, stream>>>(Qh, Kh, Vt, dsum, resT);
  k_final<<<B_ * 128 * 12 / 8, 256, 0, stream>>>(resT, Wpt, bp, x, out);
}
